// SpGAT_4329327034639
// MI455X (gfx1250) — matrix operand order repaired and checked by reading, not compiled, not run on hardware
//
#include <hip/hip_runtime.h>
#include <hip/hip_bf16.h>

// ---------------------------------------------------------------------------
// SpGAT (KBGAT) forward for MI455X / gfx1250.
//  - bf16 WMMA (v_wmma_f32_16x16x32_bf16) for the per-edge gather-GEMMs
//  - node features stored bf16 -> halves gather bandwidth (memory-bound)
//  - 4-wave workgroups share one staged edge tile (occupancy ~48 waves/WGP)
//  - global_load_async_to_lds_b128 for the pure bf16 row gathers (ASYNCcnt)
//  - global f32 atomics for segment sums (accumulators L2-resident, 51 MB)
// ---------------------------------------------------------------------------

#define NNODES 50000
#define NRELS  200
#define E1N    200000
#define E2N    100000
#define ETOT   (E1N + E2N)      // 300000, divisible by 16
#define NHID1  128              // per-head width (layer 1)
#define NHID2  256              // layer 2 width

#define USE_ASYNC_LDS 1

typedef __attribute__((ext_vector_type(16))) __bf16 v16bf;
typedef __attribute__((ext_vector_type(8)))  float  v8f;

__device__ __forceinline__ unsigned short f2bf(float x) {
    unsigned int u = __float_as_uint(x);
    u += 0x7FFFu + ((u >> 16) & 1u);      // round-to-nearest-even
    return (unsigned short)(u >> 16);
}

__device__ __forceinline__ v16bf mk16(uint4 lo, uint4 hi) {
    union { uint4 u[2]; v16bf v; } t;
    t.u[0] = lo; t.u[1] = hi;
    return t.v;
}

// 16B global -> LDS copy. Async variant lands directly in LDS (ASYNCcnt).
__device__ __forceinline__ void copy_b128_g2lds(unsigned short* lds_dst,
                                                const unsigned short* gsrc) {
#if USE_ASYNC_LDS
    unsigned int lds_off = (unsigned int)(unsigned long long)lds_dst;
    asm volatile("global_load_async_to_lds_b128 %0, %1, off"
                 :: "v"(lds_off), "v"(gsrc) : "memory");
#else
    *(uint4*)lds_dst = *(const uint4*)gsrc;
#endif
}

__device__ __forceinline__ void async_fence() {
#if USE_ASYNC_LDS
    asm volatile("s_wait_asynccnt 0x0" ::: "memory");
#endif
}

// ---------------------------------------------------------------------------
// Utility kernels
// ---------------------------------------------------------------------------
__global__ void zero_f32_kernel(float* __restrict__ p, int n) {
    int i = blockIdx.x * blockDim.x + threadIdx.x;
    if (i < n) p[i] = 0.0f;
}

__global__ void cvt_f32_bf16_kernel(const float* __restrict__ src,
                                    unsigned short* __restrict__ dst, int n) {
    int i = blockIdx.x * blockDim.x + threadIdx.x;
    if (i < n) dst[i] = f2bf(src[i]);
}

// out_relation_1 = relation_embed [200,128] @ W [128,256]  (f32, tiny)
__global__ void relout_kernel(const float* __restrict__ rel,
                              const float* __restrict__ W,
                              float* __restrict__ out) {
    int gid = blockIdx.x * blockDim.x + threadIdx.x;
    if (gid >= NRELS * NHID2) return;
    int r = gid / NHID2, c = gid % NHID2;
    float acc = 0.0f;
    #pragma unroll 4
    for (int k = 0; k < 128; ++k)
        acc = fmaf(rel[r * 128 + k], W[k * NHID2 + c], acc);
    out[gid] = acc;
}

// layer-1 finalize: x2_bf16[n, h*128+j] = bf16(elu(accum[h][n][j] / rowsum[h][n]))
__global__ void finalize1_kernel(const float* __restrict__ accum,
                                 const float* __restrict__ rowsum,
                                 unsigned short* __restrict__ x2bf) {
    int gid = blockIdx.x * blockDim.x + threadIdx.x;
    if (gid >= NNODES * NHID2) return;
    int n = gid / NHID2, j = gid % NHID2;
    int h = j >> 7, jj = j & 127;
    float rs = rowsum[h * NNODES + n];
    if (rs == 0.0f) rs = 1e-12f;
    float v = accum[((size_t)h * NNODES + n) * NHID1 + jj] / rs;
    x2bf[gid] = f2bf(v > 0.0f ? v : expm1f(v));
}

// layer-2 finalize: out[n, j] = elu(accum[n][j] / rowsum[n])   (f32 output)
__global__ void finalize2_kernel(const float* __restrict__ accum,
                                 const float* __restrict__ rowsum,
                                 float* __restrict__ out) {
    int gid = blockIdx.x * blockDim.x + threadIdx.x;
    if (gid >= NNODES * NHID2) return;
    int n = gid / NHID2;
    float rs = rowsum[n];
    if (rs == 0.0f) rs = 1e-12f;
    float v = accum[gid] / rs;
    out[gid] = v > 0.0f ? v : expm1f(v);
}

// ---------------------------------------------------------------------------
// Edge attention kernel: 128 threads (4 wave32) per 16-edge tile.
// Waves split the INF/16 output n-tiles; D tiles stay in registers.
//   INF   : node-feature width == output width (128 layer1, 256 layer2)
//   LAYER : 1 -> emb = edge_embed[e] (f32) or rel[t0]+rel[t1] (f32 sum)
//           2 -> emb = outrel_bf16[edge_type[e]] or outrel[t0]+outrel[t1]
// ---------------------------------------------------------------------------
template <int INF, int LAYER>
__global__ __launch_bounds__(128)
void gat_edge_kernel(const unsigned short* __restrict__ xbf,   // [N, INF] bf16
                     const int* __restrict__ el,               // [2, E1]
                     const int* __restrict__ elnh,             // [2, E2]
                     const int* __restrict__ etype,            // [E1]   (layer2)
                     const int* __restrict__ etypenh,          // [E2,2]
                     const float* __restrict__ embf32,         // [E1, INF] f32 (layer1)
                     const float* __restrict__ relf32,         // [R, INF] f32
                     const unsigned short* __restrict__ relbf, // [R, INF] bf16 (layer2)
                     const unsigned short* __restrict__ aBf,   // [INF, 3*INF] bf16
                     const float* __restrict__ a2,             // [INF]
                     float* __restrict__ accum,                // [N, INF]
                     float* __restrict__ rowsum)               // [N]
{
    constexpr int INF3 = 3 * INF;
    constexpr int NT   = INF / 16;     // output n-tiles (8 or 16)
    constexpr int NTW  = NT / 4;       // n-tiles per wave (2 or 4)
    constexpr int C8   = INF / 8;      // 8-element chunks per segment row

    __shared__ unsigned short Flds[16 * INF3];   // bf16 feature tile [16][3*INF]
    __shared__ float score[16];                  // edge_m @ a2 (full sum)
    __shared__ float fvals[16];                  // exp(-lrelu(score))
    __shared__ int srcl[16];
    __shared__ int dstl[16];

    const int tid  = threadIdx.x;
    const int lane = tid & 31;
    const int wid  = tid >> 5;
    const int half = lane >> 4;
    const int col  = lane & 15;
    const int eb   = blockIdx.x * 16;

    // ---- stage edge endpoints, zero score -------------------------------
    if (tid < 16) {
        int e = eb + tid;
        int s, d;
        if (e < E1N) { s = el[e];            d = el[E1N + e]; }
        else { int en = e - E1N; s = elnh[en]; d = elnh[E2N + en]; }
        srcl[tid]  = s;
        dstl[tid]  = d;
        score[tid] = 0.0f;
    }
    __syncthreads();

    // ---- stage [16][3*INF] bf16 tile in LDS (16B chunks) ---------------
    for (int idx = tid; idx < 16 * 3 * C8; idx += 128) {
        int eL  = idx / (3 * C8);
        int rem = idx % (3 * C8);
        int seg = rem / C8;
        int o8  = (rem % C8) * 8;                 // element offset in segment
        unsigned short* dst = Flds + eL * INF3 + seg * INF + o8;
        if (seg < 2) {
            int node = (seg == 0) ? srcl[eL] : dstl[eL];
            copy_b128_g2lds(dst, xbf + (size_t)node * INF + o8);      // async
        } else {
            int e = eb + eL;
            if (e < E1N) {
                if (LAYER == 2) {
                    int t = etype[e];
                    copy_b128_g2lds(dst, relbf + (size_t)t * INF + o8); // async
                } else {
                    const float* sp = embf32 + (size_t)e * INF + o8;
                    float4 v0 = *(const float4*)(sp);
                    float4 v1 = *(const float4*)(sp + 4);
                    ushort4 b0, b1;
                    b0.x = f2bf(v0.x); b0.y = f2bf(v0.y); b0.z = f2bf(v0.z); b0.w = f2bf(v0.w);
                    b1.x = f2bf(v1.x); b1.y = f2bf(v1.y); b1.z = f2bf(v1.z); b1.w = f2bf(v1.w);
                    *(ushort4*)dst       = b0;
                    *(ushort4*)(dst + 4) = b1;
                }
            } else {
                int en = e - E1N;
                int t0 = etypenh[2 * en], t1 = etypenh[2 * en + 1];
                const float* p0 = relf32 + (size_t)t0 * INF + o8;
                const float* p1 = relf32 + (size_t)t1 * INF + o8;
                float4 a0 = *(const float4*)(p0),     a1 = *(const float4*)(p0 + 4);
                float4 c0 = *(const float4*)(p1),     c1 = *(const float4*)(p1 + 4);
                ushort4 b0, b1;
                b0.x = f2bf(a0.x + c0.x); b0.y = f2bf(a0.y + c0.y);
                b0.z = f2bf(a0.z + c0.z); b0.w = f2bf(a0.w + c0.w);
                b1.x = f2bf(a1.x + c1.x); b1.y = f2bf(a1.y + c1.y);
                b1.z = f2bf(a1.z + c1.z); b1.w = f2bf(a1.w + c1.w);
                *(ushort4*)dst       = b0;
                *(ushort4*)(dst + 4) = b1;
            }
        }
    }
    async_fence();                 // this wave's async LDS writes complete
    __syncthreads();               // tile visible to all 4 waves

    // ---- WMMA: each wave computes NTW n-tiles, D stays in registers -----
    const unsigned short* Fa = Flds + (size_t)col * INF3;  // A row m = col

    v8f   ctile[NTW];
    float part[8];
    #pragma unroll
    for (int r = 0; r < 8; ++r) part[r] = 0.0f;

    #pragma unroll
    for (int t = 0; t < NTW; ++t) {
        const int nt = wid * NTW + t;
        const int n  = nt * 16 + col;
        const unsigned short* Bn = aBf + (size_t)n * INF3;
        const float a2v = a2[n];

        v8f c = {};
        #pragma unroll 4
        for (int kc = 0; kc < INF3; kc += 32) {
            // A 16x32 bf16: lanes 0-15 -> K kc+0..7/kc+16..23, lanes 16-31 -> +8
            const unsigned short* ap = Fa + kc + half * 8;
            uint4 a0 = *(const uint4*)(ap);
            uint4 a1 = *(const uint4*)(ap + 16);
            // B 32x16 bf16: lane = column n, lanes 0-15 -> K kc+0..7/kc+16..23, lanes 16-31 -> +8
            const unsigned short* bp = Bn + kc + half * 8;
            uint4 b0 = *(const uint4*)(bp);
            uint4 b1 = *(const uint4*)(bp + 16);
            c = __builtin_amdgcn_wmma_f32_16x16x32_bf16(
                    false, mk16(a0, a1), false, mk16(b0, b1),
                    (short)0, c, false, false);
        }
        ctile[t] = c;
        // C layout: reg r at lane l -> m = r + 8*half, n = nt*16 + col
        #pragma unroll
        for (int r = 0; r < 8; ++r)
            part[r] = fmaf(c[r], a2v, part[r]);
    }

    // ---- a2-dot: butterfly within 16-lane halves, then LDS combine ------
    #pragma unroll
    for (int r = 0; r < 8; ++r) {
        float s = part[r];
        s += __shfl_xor(s, 1);
        s += __shfl_xor(s, 2);
        s += __shfl_xor(s, 4);
        s += __shfl_xor(s, 8);
        if (col == 0) atomicAdd(&score[r + 8 * half], s);   // ds_add_f32
    }
    __syncthreads();

    // ---- edge_e = exp(-leaky_relu(score)), rowsum atomic ---------------
    if (tid < 16) {
        float s  = score[tid];
        float lr = s >= 0.0f ? s : 0.2f * s;
        float f  = expf(-lr);
        fvals[tid] = f;
        atomicAdd(&rowsum[srcl[tid]], f);
    }
    __syncthreads();

    // ---- scatter edge_e * edge_m from registers -------------------------
    float fr[8];
    int   nd[8];
    #pragma unroll
    for (int r = 0; r < 8; ++r) {
        int m = r + 8 * half;
        fr[r] = fvals[m];
        nd[r] = srcl[m];
    }
    #pragma unroll
    for (int t = 0; t < NTW; ++t) {
        const int n = (wid * NTW + t) * 16 + col;
        #pragma unroll
        for (int r = 0; r < 8; ++r)
            atomicAdd(&accum[(size_t)nd[r] * INF + n], fr[r] * ctile[t][r]);
    }
}

// ---------------------------------------------------------------------------
// Host-side launch
// ---------------------------------------------------------------------------
extern "C" void kernel_launch(void* const* d_in, const int* in_sizes, int n_in,
                              void* d_out, int out_size, void* d_ws, size_t ws_size,
                              hipStream_t stream) {
    (void)in_sizes; (void)n_in; (void)out_size; (void)ws_size;

    const float* ent   = (const float*)d_in[0];   // [50000,128]
    const float* rel   = (const float*)d_in[1];   // [200,128]
    const float* eemb  = (const float*)d_in[2];   // [200000,128]
    const float* aH    = (const float*)d_in[3];   // [2,128,384]
    const float* a2H   = (const float*)d_in[4];   // [2,128]
    const float* aO    = (const float*)d_in[5];   // [256,768]
    const float* a2O   = (const float*)d_in[6];   // [256]
    const float* W     = (const float*)d_in[7];   // [128,256]
    const int*   el    = (const int*)d_in[8];     // [2,200000]
    const int*   et    = (const int*)d_in[9];     // [200000]
    const int*   elnh  = (const int*)d_in[10];    // [2,100000]
    const int*   etnh  = (const int*)d_in[11];    // [100000,2]

    float* out    = (float*)d_out;                       // [50000,256] then [200,256]
    float* outrel = out + (size_t)NNODES * NHID2;        // out_relation_1 slot

    const int NACC = NNODES * NHID2;                     // 12.8M

    // workspace layout (all offsets 16B-aligned)
    float* accum   = (float*)d_ws;                       // [12.8M] (reused L1/L2)
    float* rowsum1 = accum + NACC;                       // [2*50000]
    float* rowsum2 = rowsum1 + 2 * NNODES;               // [50000]
    unsigned short* aHbf  = (unsigned short*)(rowsum2 + NNODES); // [2*128*384]
    unsigned short* aObf  = aHbf + 2 * 128 * 384;                // [256*768]
    unsigned short* xbf   = aObf + 256 * 768;                    // [50000*128]
    unsigned short* x2bf  = xbf + NNODES * NHID1;                // [50000*256]
    unsigned short* orlbf = x2bf + NACC;                         // [200*256]

    const int TB    = 256;
    const int NEBLK = ETOT / 16;                         // 18750

    // 1) zero layer-1 accumulators + both rowsums
    zero_f32_kernel<<<(NACC + TB - 1) / TB, TB, 0, stream>>>(accum, NACC);
    zero_f32_kernel<<<(3 * NNODES + TB - 1) / TB, TB, 0, stream>>>(rowsum1, 3 * NNODES);

    // 2) bf16 copies: attention matrices + node features
    cvt_f32_bf16_kernel<<<(2 * 128 * 384 + TB - 1) / TB, TB, 0, stream>>>(aH, aHbf, 2 * 128 * 384);
    cvt_f32_bf16_kernel<<<(256 * 768 + TB - 1) / TB, TB, 0, stream>>>(aO, aObf, 256 * 768);
    cvt_f32_bf16_kernel<<<(NNODES * NHID1 + TB - 1) / TB, TB, 0, stream>>>(ent, xbf, NNODES * NHID1);

    // 3) layer-1 attention, one launch per head (4 waves / 16-edge tile)
    gat_edge_kernel<NHID1, 1><<<NEBLK, 128, 0, stream>>>(
        xbf, el, elnh, nullptr, etnh, eemb, rel, nullptr,
        aHbf, a2H, accum, rowsum1);
    gat_edge_kernel<NHID1, 1><<<NEBLK, 128, 0, stream>>>(
        xbf, el, elnh, nullptr, etnh, eemb, rel, nullptr,
        aHbf + 128 * 384, a2H + 128,
        accum + (size_t)NNODES * NHID1, rowsum1 + NNODES);

    // 4) layer-1 finalize -> x2 (bf16, consumed only by layer 2)
    finalize1_kernel<<<(NACC + TB - 1) / TB, TB, 0, stream>>>(accum, rowsum1, x2bf);

    // 5) out_relation_1 = rel @ W (f32 output slot) + bf16 copy for gathers
    relout_kernel<<<(NRELS * NHID2 + TB - 1) / TB, TB, 0, stream>>>(rel, W, outrel);
    cvt_f32_bf16_kernel<<<(NRELS * NHID2 + TB - 1) / TB, TB, 0, stream>>>(outrel, orlbf, NRELS * NHID2);

    // 6) zero layer-2 accumulator (rowsum2 already zeroed in step 1)
    zero_f32_kernel<<<(NACC + TB - 1) / TB, TB, 0, stream>>>(accum, NACC);

    // 7) layer-2 attention
    gat_edge_kernel<NHID2, 2><<<NEBLK, 128, 0, stream>>>(
        x2bf, el, elnh, et, etnh, nullptr, outrel, orlbf,
        aObf, a2O, accum, rowsum2);

    // 8) layer-2 finalize -> out [50000,256]
    finalize2_kernel<<<(NACC + TB - 1) / TB, TB, 0, stream>>>(accum, rowsum2, out);
}
